// XNORConv1d_75170517614964
// MI455X (gfx1250) — 8-bit matrix operand order repaired and checked by reading, compiled, not run on hardware
//
#include <hip/hip_runtime.h>

typedef __attribute__((ext_vector_type(8))) int v8i;

#define CIN    512
#define COUT   512
#define LEN    4096
#define KW     3
#define NBATCH 32

#define XPITCH 144
#define APITCH 144

#define WS_PARTIAL 0
#define WS_SCALEX  2048
#define WS_SCALEW  2080

__global__ __launch_bounds__(256) void absmean_x_partial(const float* __restrict__ x,
                                                         float* __restrict__ ws) {
  const int b = blockIdx.y, p = blockIdx.x, t = threadIdx.x;
  const float4* xb = (const float4*)(x + (size_t)b * CIN * LEN);
  size_t base = (size_t)p * 8192 + t;
  float s = 0.f;
#pragma unroll 4
  for (int i = 0; i < 32; ++i) {
    float4 v = xb[base + (size_t)i * 256];
    s += fabsf(v.x) + fabsf(v.y) + fabsf(v.z) + fabsf(v.w);
  }
  __shared__ float red[256];
  red[t] = s; __syncthreads();
  for (int off = 128; off > 0; off >>= 1) {
    if (t < off) red[t] += red[t + off];
    __syncthreads();
  }
  if (t == 0) ws[WS_PARTIAL + b * 64 + p] = red[0];
}

__global__ __launch_bounds__(64) void absmean_x_final(float* __restrict__ ws) {
  const int b = blockIdx.x, t = threadIdx.x;
  __shared__ float red[64];
  red[t] = ws[WS_PARTIAL + b * 64 + t]; __syncthreads();
  for (int off = 32; off > 0; off >>= 1) {
    if (t < off) red[t] += red[t + off];
    __syncthreads();
  }
  if (t == 0) ws[WS_SCALEX + b] = red[0] / (float)(CIN * LEN);
}

__global__ __launch_bounds__(256) void absmean_w(const float* __restrict__ w,
                                                 float* __restrict__ ws) {
  const int co = blockIdx.x, t = threadIdx.x;
  const float* wr = w + (size_t)co * (CIN * KW);
  float s = 0.f;
  for (int i = t; i < CIN * KW; i += 256) s += fabsf(wr[i]);
  __shared__ float red[256];
  red[t] = s; __syncthreads();
  for (int off = 128; off > 0; off >>= 1) {
    if (t < off) red[t] += red[t + off];
    __syncthreads();
  }
  if (t == 0) ws[WS_SCALEW + co] = red[0] / (float)(CIN * KW);
}

__global__ __launch_bounds__(256) void xnor_conv_wmma(const float* __restrict__ x,
                                                      const float* __restrict__ w,
                                                      const float* __restrict__ ws,
                                                      float* __restrict__ y) {
  const int l_blk  = blockIdx.x * 128;
  const int co_blk = blockIdx.y * 64;
  const int b      = blockIdx.z;
  const int t      = threadIdx.x;
  const int lane   = t & 31;
  const int wid    = t >> 5;
  const int wm     = wid & 3;
  const int wn     = wid >> 2;
  const int half   = lane >> 4;
  const int l16    = lane & 15;

  __shared__ __align__(16) signed char Xs[130 * XPITCH];
  __shared__ __align__(16) signed char As[KW * 64 * APITCH];

  v8i acc[4] = {};

  const size_t xoff = (size_t)b * CIN * LEN;

  for (int cc = 0; cc < CIN; cc += 128) {
    __syncthreads();
    for (int e = t; e < 130 * 128; e += 256) {
      int row = e % 130;
      int ci  = e / 130;
      int l   = l_blk - 1 + row;
      signed char sv = 0;
      if (l >= 0 && l < LEN) {
        float v = x[xoff + (size_t)(cc + ci) * LEN + l];
        sv = (signed char)((v > 0.f) - (v < 0.f));
      }
      Xs[row * XPITCH + ci] = sv;
    }
    for (int e = t; e < KW * 64 * 128; e += 256) {
      int ci = e & 127;
      int co = (e >> 7) & 63;
      int k  = e >> 13;
      float v = w[(size_t)(co_blk + co) * (CIN * KW) + (size_t)(cc + ci) * KW + k];
      As[(k * 64 + co) * APITCH + ci] = (signed char)((v > 0.f) - (v < 0.f));
    }
    if (cc + 128 < CIN) {
      int pr = t & 127;
      int pq = t >> 7;
#pragma unroll
      for (int i = 0; i < 2; ++i) {
        int lo = l_blk + (pq * 2 + i) * 32;
        __builtin_prefetch(&x[xoff + (size_t)(cc + 128 + pr) * LEN + lo], 0, 3);
      }
    }
    __syncthreads();

    for (int k = 0; k < KW; ++k) {
#pragma unroll
      for (int c = 0; c < 2; ++c) {
        v8i a;
        const signed char* arow = &As[(k * 64 + wm * 16 + l16) * APITCH + c * 64];
#pragma unroll
        for (int v = 0; v < 8; ++v) {
          int koff = (half ? 8 : 0) + ((v & 1) * 4) + (((v >> 1) & 1) * 16) + ((v >> 2) * 32);
          a[v] = *reinterpret_cast<const int*>(arow + koff);
        }
#pragma unroll
        for (int tn = 0; tn < 4; ++tn) {
          int col = wn * 64 + tn * 16 + l16;
          const signed char* brow = &Xs[(col + k) * XPITCH + c * 64];
          v8i bf;
#pragma unroll
          for (int v = 0; v < 8; ++v) {
            int koff = (half ? 8 : 0) + ((v & 1) * 4) + (((v >> 1) & 1) * 16) + ((v >> 2) * 32);
            bf[v] = *reinterpret_cast<const int*>(brow + koff);
          }
          acc[tn] = __builtin_amdgcn_wmma_i32_16x16x64_iu8(
               true, a,  true, bf, acc[tn],
               false,  false);
        }
      }
    }
  }

  const float sx = ws[WS_SCALEX + b];
#pragma unroll
  for (int tn = 0; tn < 4; ++tn) {
#pragma unroll
    for (int j = 0; j < 8; ++j) {
      int co = co_blk + wm * 16 + half * 8 + j;
      int l  = l_blk + wn * 64 + tn * 16 + l16;
      float sw = ws[WS_SCALEW + co];
      y[((size_t)b * COUT + co) * LEN + l] = (float)acc[tn][j] * sx * sw;
    }
  }
}

extern "C" void kernel_launch(void* const* d_in, const int* in_sizes, int n_in,
                              void* d_out, int out_size, void* d_ws, size_t ws_size,
                              hipStream_t stream) {
  const float* x = (const float*)d_in[0];
  const float* w = (const float*)d_in[1];
  float* ws = (float*)d_ws;
  float* y  = (float*)d_out;

  absmean_x_partial<<<dim3(64, NBATCH), 256, 0, stream>>>(x, ws);
  absmean_x_final<<<NBATCH, 64, 0, stream>>>(ws);
  absmean_w<<<COUT, 256, 0, stream>>>(w, ws);
  xnor_conv_wmma<<<dim3(LEN / 128, COUT / 64, NBATCH), 256, 0, stream>>>(x, w, ws, y);
}
